// LidarBEVDetection_23759759082222
// MI455X (gfx1250) — matrix operand order repaired and checked by reading, not compiled, not run on hardware
//
#include <hip/hip_runtime.h>
#include <hip/hip_bf16.h>

typedef __attribute__((ext_vector_type(16))) __bf16 v16bf;
typedef __attribute__((ext_vector_type(8)))  __bf16 v8bf;
typedef __attribute__((ext_vector_type(8)))  float  v8f;

#define HH 512
#define WW 512
#define HW (HH*WW)
#define CC 128
#define DD 256
#define KTOP 500
#define MP 512          // padded M for transformer GEMMs
#define CAP 4096

__device__ __forceinline__ v8f wmma_bf16(v16bf a, v16bf b, v8f c) {
  return __builtin_amdgcn_wmma_f32_16x16x32_bf16(false, a, false, b, (short)0, c, false, false);
}

// ---------------- utility ----------------
__global__ void zero_u32(unsigned* p, int n) {
  int i = blockIdx.x * blockDim.x + threadIdx.x;
  if (i < n) p[i] = 0u;
}

// zero-padded row copy: [rows x 256] -> [512 x 256]
__global__ void pad_copy(const float* __restrict__ src, float* __restrict__ dst, int rows) {
  int k = blockIdx.x, c = threadIdx.x;   // 512 x 256
  dst[(long long)k * DD + c] = (k < rows) ? src[(long long)k * DD + c] : 0.0f;
}

// transpose [512 x 256] -> [256 x 512]
__global__ void transpose_kt(const float* __restrict__ src, float* __restrict__ dst) {
  int r = blockIdx.x, c = threadIdx.x;   // 512 x 256
  dst[(long long)c * MP + r] = src[(long long)r * DD + c];
}

// ---------------- NMS + histogram (radix-select level 0) ----------------
__global__ void nms_hist(const float* __restrict__ hm, unsigned* __restrict__ keys,
                         unsigned* __restrict__ hist) {
  int i = blockIdx.x * blockDim.x + threadIdx.x;   // HW threads
  int h = i >> 9, w = i & 511;
  float c = hm[i];
  float m = c;
  #pragma unroll
  for (int dy = -1; dy <= 1; ++dy) {
    int r = h + dy;
    if (r < 0 || r >= HH) continue;
    #pragma unroll
    for (int dx = -1; dx <= 1; ++dx) {
      int cw = w + dx;
      if (cw < 0 || cw >= WW) continue;
      m = fmaxf(m, hm[r * WW + cw]);
    }
  }
  float s = (c == m) ? c : 0.0f;
  unsigned k = __float_as_uint(s);   // s >= 0 -> bits order preserving
  keys[i] = k;
  atomicAdd(&hist[k >> 20], 1u);
}

__global__ void select_thresh(const unsigned* __restrict__ hist, unsigned* sel) {
  if (threadIdx.x == 0 && blockIdx.x == 0) {
    unsigned acc = 0, B = 0;
    for (int b = 4095; b >= 0; --b) {
      acc += hist[b];
      if (acc >= (unsigned)KTOP) { B = (unsigned)b; break; }
    }
    sel[0] = B;
  }
}

__global__ void compact_cand(const unsigned* __restrict__ keys, const unsigned* __restrict__ sel,
                             unsigned* counter, unsigned long long* cand) {
  int i = blockIdx.x * blockDim.x + threadIdx.x;
  unsigned k = keys[i];
  if ((k >> 20) >= sel[0]) {
    unsigned pos = atomicAdd(counter, 1u);
    if (pos < (unsigned)CAP)
      cand[pos] = ((unsigned long long)k << 32) | (unsigned)(~(unsigned)i);
  }
}

// single-block bitonic sort (descending) of up to CAP keys, emit top-K coords
__global__ void sort_emit(const unsigned long long* __restrict__ cand,
                          const unsigned* __restrict__ counter, int* xs, int* ys) {
  __shared__ unsigned long long sk[CAP];
  int t = threadIdx.x;     // 1024 threads
  unsigned n = *counter; if (n > (unsigned)CAP) n = CAP;
  for (int i = t; i < CAP; i += 1024) sk[i] = (i < (int)n) ? cand[i] : 0ULL;
  __syncthreads();
  for (int k = 2; k <= CAP; k <<= 1) {
    for (int j = k >> 1; j > 0; j >>= 1) {
      for (int i = t; i < CAP; i += 1024) {
        int ixj = i ^ j;
        if (ixj > i) {
          unsigned long long a = sk[i], b = sk[ixj];
          bool desc = ((i & k) == 0);
          if (desc ? (a < b) : (a > b)) { sk[i] = b; sk[ixj] = a; }
        }
      }
      __syncthreads();
    }
  }
  if (t < KTOP) {
    unsigned long long kk = sk[t];
    unsigned idx = ((unsigned)(kk >> 32) != 0u) ? ~(unsigned)(kk & 0xFFFFFFFFu) : 0u;
    xs[t] = (int)(idx >> 9);
    ys[t] = (int)(idx & 511u);
  }
}

// ---------------- gather features + sine positional embedding ----------------
__global__ void gather_feat(const float* __restrict__ fm, const int* __restrict__ xs,
                            const int* __restrict__ ys, float* __restrict__ AF) {
  int k = blockIdx.x;        // 512 blocks (rows >= KTOP zeroed for padding)
  int c = threadIdx.x;       // 256
  float v = 0.0f;
  if (k < KTOP) {
    int x = xs[k], y = ys[k];
    if (c < CC) {
      v = fm[(long long)c * HW + x * WW + y];
    } else {
      int cc = c - CC;
      int j; float coord;
      if (cc < 64) { j = cc >> 1;        coord = (float)(x + 1); }   // py (row)
      else         { j = (cc - 64) >> 1; coord = (float)(y + 1); }   // px (col)
      float tt = powf(10000.0f, (float)j * (1.0f / 32.0f));
      float ang = coord / tt;
      v = (cc & 1) ? cosf(ang) : sinf(ang);
    }
  }
  AF[(long long)k * DD + c] = v;
}

// ---------------- guard-free batched bf16-WMMA GEMM ----------------
// Requires: M%16==0, N%16==0, Kd%32==0, lda/ldb/ldc multiples of 4.
// C = alpha*A*B (+bias) (relu)
__global__ __launch_bounds__(256) void gemm_wmma(
    const float* __restrict__ A, int lda, long long sA,
    const float* __restrict__ B, int ldb, long long sB,
    float* __restrict__ C, int ldc, long long sC,
    int M, int N, int Kd, float alpha,
    const float* __restrict__ bias, int relu) {
  int lane = threadIdx.x & 31;
  int wid  = threadIdx.x >> 5;
  int Mt = M >> 4, Nt = N >> 4;
  int tile = blockIdx.x * 8 + wid;
  if (tile >= Mt * Nt) return;
  int mt = tile / Nt, nt = tile % Nt;
  int laneHi = (lane & 16) ? 8 : 0;
  int row = mt * 16 + (lane & 15);
  const float* Arow = A + (long long)blockIdx.y * sA + (long long)row * lda;
  const float* Bbase = B + (long long)blockIdx.y * sB + nt * 16;
  v8f acc = {0.f,0.f,0.f,0.f,0.f,0.f,0.f,0.f};
  for (int kb = 0; kb < Kd; kb += 32) {
    // A fragment: two contiguous 8-float runs at kb+laneHi and kb+16+laneHi
    const float4* pa = (const float4*)(Arow + kb + laneHi);
    const float4* pb = (const float4*)(Arow + kb + 16 + laneHi);
    float4 a0 = pa[0], a1 = pa[1], a2 = pb[0], a3 = pb[1];
    v16bf af;
    af[0]=(__bf16)a0.x;  af[1]=(__bf16)a0.y;  af[2]=(__bf16)a0.z;  af[3]=(__bf16)a0.w;
    af[4]=(__bf16)a1.x;  af[5]=(__bf16)a1.y;  af[6]=(__bf16)a1.z;  af[7]=(__bf16)a1.w;
    af[8]=(__bf16)a2.x;  af[9]=(__bf16)a2.y;  af[10]=(__bf16)a2.z; af[11]=(__bf16)a2.w;
    af[12]=(__bf16)a3.x; af[13]=(__bf16)a3.y; af[14]=(__bf16)a3.z; af[15]=(__bf16)a3.w;
    // B fragment: lane = N column, K rows kb+laneHi+0..7 and kb+16+laneHi+0..7
    const float* pq = Bbase + (lane & 15) + (long long)(kb + laneHi) * ldb;
    v16bf bf;
    #pragma unroll
    for (int e = 0; e < 8; ++e) {
      bf[e]     = (__bf16)pq[(long long)e * ldb];
      bf[e + 8] = (__bf16)pq[(long long)(e + 16) * ldb];
    }
    acc = wmma_bf16(af, bf, acc);
  }
  int n = nt * 16 + (lane & 15);
  float bb = (bias != nullptr) ? bias[n] : 0.0f;
  float* Crow = C + (long long)blockIdx.y * sC + n;
  #pragma unroll
  for (int r = 0; r < 8; ++r) {
    int m = mt * 16 + r + laneHi;
    float v = alpha * acc[r] + bb;
    if (relu) v = fmaxf(v, 0.0f);
    Crow[(long long)m * ldc] = v;
  }
}

// ---------------- row softmax (per head, per row) ----------------
__global__ void softmax_rows(float* __restrict__ S, int ld, int cols, long long stride) {
  int row = blockIdx.x, head = blockIdx.y, t = threadIdx.x;
  float* p = S + (long long)head * stride + (long long)row * ld;
  __shared__ float red[256];
  float mx = -3.0e38f;
  for (int i = t; i < cols; i += 256) mx = fmaxf(mx, p[i]);
  red[t] = mx; __syncthreads();
  for (int s = 128; s > 0; s >>= 1) { if (t < s) red[t] = fmaxf(red[t], red[t + s]); __syncthreads(); }
  mx = red[0]; __syncthreads();
  float sum = 0.0f;
  for (int i = t; i < cols; i += 256) { float e = expf(p[i] - mx); p[i] = e; sum += e; }
  red[t] = sum; __syncthreads();
  for (int s = 128; s > 0; s >>= 1) { if (t < s) red[t] += red[t + s]; __syncthreads(); }
  float inv = 1.0f / red[0];
  for (int i = t; i < cols; i += 256) p[i] *= inv;
}

// ---------------- residual + LayerNorm over D=256 ----------------
__global__ void ln_kernel(const float* __restrict__ a, const float* __restrict__ resid,
                          const float* __restrict__ g, const float* __restrict__ b,
                          float* __restrict__ o) {
  int row = blockIdx.x, t = threadIdx.x;   // 256 threads == D
  long long off = (long long)row * DD + t;
  float x = a[off] + (resid ? resid[off] : 0.0f);
  __shared__ float red[256];
  red[t] = x; __syncthreads();
  for (int s = 128; s > 0; s >>= 1) { if (t < s) red[t] += red[t + s]; __syncthreads(); }
  float mean = red[0] * (1.0f / DD); __syncthreads();
  float d = x - mean;
  red[t] = d * d; __syncthreads();
  for (int s = 128; s > 0; s >>= 1) { if (t < s) red[t] += red[t + s]; __syncthreads(); }
  float var = red[0] * (1.0f / DD);
  o[off] = d * rsqrtf(var + 1e-5f) * g[t] + b[t];
}

// ---------------- scatter refined features back into BEV copy ----------------
__global__ void scatter_kernel(float* __restrict__ fmc, const int* __restrict__ xs,
                               const int* __restrict__ ys, const float* __restrict__ fin) {
  int k = blockIdx.x, c = threadIdx.x;   // 500 x 128
  fmc[(long long)c * HW + xs[k] * WW + ys[k]] = fin[(long long)k * DD + c];
}

// ---------------- pack conv weights OIHW f32 -> [64][1152] bf16 (k = tap*128 + c) ----------------
__global__ void pack_w(const float* __restrict__ w, __bf16* __restrict__ out) {
  int i = blockIdx.x * blockDim.x + threadIdx.x;
  if (i >= 64 * 1152) return;
  int o = i / 1152, k = i % 1152;
  int t = k >> 7, c = k & 127;
  int ky = t / 3, kx = t % 3;
  out[i] = (__bf16)w[((o * CC + c) * 3 + ky) * 3 + kx];
}

// ---------------- fused conv3x3(128->64)+bias+relu+conv1x1(64->2)+bias ----------------
// Implicit GEMM: M=64 outch (4 tiles/wave), N=16 pixels/wave, K=1152.
// All control flow is wave-uniform; all loads unguarded & aligned in the interior.
__global__ __launch_bounds__(256) void conv_fused(
    const float* __restrict__ fm, const __bf16* __restrict__ Wpk,
    const float* __restrict__ b1, const float* __restrict__ w2,
    const float* __restrict__ b2, float* __restrict__ out) {
  int lane = threadIdx.x & 31, wid = threadIdx.x >> 5;
  int p = blockIdx.x * 8 + wid;           // 16384 pixel-tiles (16 px each)
  int h = p >> 5, wb = (p & 31) << 4;     // wave-uniform
  int laneHi = (lane & 16) ? 8 : 0;
  int arow = lane & 15;
  bool ledge = (wb == 0), redge = (wb == WW - 16);
  v8f z = {0.f,0.f,0.f,0.f,0.f,0.f,0.f,0.f};
  v8f acc[4] = {z, z, z, z};
  #pragma unroll
  for (int dyi = 0; dyi < 3; ++dyi) {
    int sr = h + dyi - 1;
    bool rok = (sr >= 0 && sr < HH);      // wave-uniform
    #pragma unroll
    for (int kb = 0; kb < 4; ++kb) {
      int n = lane & 15;
      #pragma unroll
      for (int dxi = 0; dxi < 3; ++dxi) {
        int sc = wb + n + dxi - 1;
        bool ok = rok && sc >= 0 && sc < WW;
        v16bf bf;
        #pragma unroll
        for (int e = 0; e < 16; ++e) {
          int c = kb * 32 + laneHi + (e & 7) + ((e >> 3) << 4);
          bf[e] = (__bf16)(ok ? fm[(long long)c * HW + (long long)sr * WW + sc] : 0.0f);
        }
        int kbase = (dyi * 3 + dxi) * CC + kb * 32;
        #pragma unroll
        for (int mt = 0; mt < 4; ++mt) {
          const __bf16* Ar = Wpk + (mt * 16 + arow) * 1152 + kbase + laneHi;
          v8bf lo = *(const v8bf*)(Ar);        // 16B-aligned
          v8bf hi = *(const v8bf*)(Ar + 16);
          v16bf af = __builtin_shufflevector(lo, hi, 0,1,2,3,4,5,6,7,8,9,10,11,12,13,14,15);
          acc[mt] = wmma_bf16(af, bf, acc[mt]);
        }
      }
    }
  }
  // fused 1x1 conv epilogue: each lane holds 32 of 64 channels for pixel n=lane%16
  float o0 = 0.0f, o1 = 0.0f;
  #pragma unroll
  for (int mt = 0; mt < 4; ++mt) {
    #pragma unroll
    for (int r = 0; r < 8; ++r) {
      int ch = mt * 16 + r + laneHi;
      float v = fmaxf(acc[mt][r] + b1[ch], 0.0f);
      o0 += w2[ch] * v;
      o1 += w2[64 + ch] * v;
    }
  }
  o0 += __shfl_xor(o0, 16, 32);
  o1 += __shfl_xor(o1, 16, 32);
  if (lane < 16) {
    int w = wb + lane;
    out[(long long)h * WW + w]      = o0 + b2[0];
    out[HW + (long long)h * WW + w] = o1 + b2[1];
  }
}

// ---------------- host launcher ----------------
extern "C" void kernel_launch(void* const* d_in, const int* in_sizes, int n_in,
                              void* d_out, int out_size, void* d_ws, size_t ws_size,
                              hipStream_t stream) {
  (void)in_sizes; (void)n_in; (void)out_size; (void)ws_size;
  const float* fm   = (const float*)d_in[0];
  const float* hm   = (const float*)d_in[1];
  const float* Pf   = (const float*)d_in[2];
  const float* Wq   = (const float*)d_in[3];
  const float* Wk   = (const float*)d_in[4];
  const float* Wv   = (const float*)d_in[5];
  const float* Wo   = (const float*)d_in[6];
  const float* W1   = (const float*)d_in[7];
  const float* b1   = (const float*)d_in[8];
  const float* W2   = (const float*)d_in[9];
  const float* b2   = (const float*)d_in[10];
  const float* ln1g = (const float*)d_in[11];
  const float* ln1b = (const float*)d_in[12];
  const float* ln2g = (const float*)d_in[13];
  const float* ln2b = (const float*)d_in[14];
  const float* whw1 = (const float*)d_in[15];
  const float* whb1 = (const float*)d_in[16];
  const float* whw2 = (const float*)d_in[17];
  const float* whb2 = (const float*)d_in[18];
  const float* orw1 = (const float*)d_in[19];
  const float* orb1 = (const float*)d_in[20];
  const float* orw2 = (const float*)d_in[21];
  const float* orb2 = (const float*)d_in[22];

  float* out = (float*)d_out;
  char* ws = (char*)d_ws;
  size_t cur = 0;
  auto alloc = [&](size_t bytes) -> char* {
    cur = (cur + 255) & ~(size_t)255;
    char* p = ws + cur;
    cur += bytes;
    return p;
  };

  unsigned*            keys = (unsigned*)alloc((size_t)HW * 4);
  unsigned*            hist = (unsigned*)alloc(4096 * 4);
  unsigned*            sel  = (unsigned*)alloc(16 * 4);          // sel[0]=bin, sel[1]=counter
  unsigned long long*  cand = (unsigned long long*)alloc((size_t)CAP * 8);
  int*                 xs   = (int*)alloc(512 * 4);
  int*                 ys   = (int*)alloc(512 * 4);
  float*               AF   = (float*)alloc((size_t)MP * DD * 4);
  float*               Pp   = (float*)alloc((size_t)MP * DD * 4);   // padded pre_topk
  float*               Qb   = (float*)alloc((size_t)MP * DD * 4);
  float*               Kb   = (float*)alloc((size_t)MP * DD * 4);
  float*               KbT  = (float*)alloc((size_t)DD * MP * 4);   // K transposed
  float*               Vb   = (float*)alloc((size_t)MP * DD * 4);
  float*               Sb   = (float*)alloc((size_t)8 * MP * MP * 4);
  float*               Ob   = (float*)alloc((size_t)MP * DD * 4);
  float*               AttO = (float*)alloc((size_t)MP * DD * 4);
  float*               X1   = (float*)alloc((size_t)MP * DD * 4);
  float*               H1   = (float*)alloc((size_t)MP * 1024 * 4);
  float*               H2   = (float*)alloc((size_t)MP * DD * 4);
  __bf16*              WpkA = (__bf16*)alloc((size_t)64 * 1152 * 2);
  __bf16*              WpkB = (__bf16*)alloc((size_t)64 * 1152 * 2);
  float*               fmc  = (float*)alloc((size_t)CC * HW * 4);

  float* out_final = out;                    // [500,256]
  float* out_hm    = out + 128000;           // [1,1,512,512]
  float* out_wh    = out + 128000 + HW;      // [1,2,512,512]
  float* out_ori   = out + 128000 + HW + 2 * HW;

  // passthrough heatmap + writable copy of feature map
  (void)hipMemcpyAsync(out_hm, hm, (size_t)HW * 4, hipMemcpyDeviceToDevice, stream);
  (void)hipMemcpyAsync(fmc, fm, (size_t)CC * HW * 4, hipMemcpyDeviceToDevice, stream);

  // ---- top-K NMS (radix-select + bitonic) ----
  zero_u32<<<17, 256, 0, stream>>>(hist, 4096);
  zero_u32<<<1, 16, 0, stream>>>(sel, 16);
  nms_hist<<<HW / 256, 256, 0, stream>>>(hm, keys, hist);
  select_thresh<<<1, 1, 0, stream>>>(hist, sel);
  compact_cand<<<HW / 256, 256, 0, stream>>>(keys, sel, sel + 1, cand);
  sort_emit<<<1, 1024, 0, stream>>>(cand, sel + 1, xs, ys);

  // ---- gather + positional embedding; pad previous-frame features ----
  gather_feat<<<MP, 256, 0, stream>>>(fm, xs, ys, AF);
  pad_copy<<<MP, 256, 0, stream>>>(Pf, Pp, KTOP);

  // ---- cross-attention (8 heads x 32), all dims padded/aligned ----
  gemm_wmma<<<dim3(64, 1), 256, 0, stream>>>(AF, DD, 0, Wq, DD, 0, Qb, DD, 0,
                                             MP, DD, DD, 1.0f, nullptr, 0);
  gemm_wmma<<<dim3(64, 1), 256, 0, stream>>>(Pp, DD, 0, Wk, DD, 0, Kb, DD, 0,
                                             MP, DD, DD, 1.0f, nullptr, 0);
  gemm_wmma<<<dim3(64, 1), 256, 0, stream>>>(Pp, DD, 0, Wv, DD, 0, Vb, DD, 0,
                                             MP, DD, DD, 1.0f, nullptr, 0);
  transpose_kt<<<MP, 256, 0, stream>>>(Kb, KbT);
  // S_h = (Q_h K_h^T)/sqrt(32): B = KbT rows are head-dim, contiguous in n
  gemm_wmma<<<dim3(128, 8), 256, 0, stream>>>(Qb, DD, 32, KbT, MP, (long long)32 * MP,
                                              Sb, MP, (long long)MP * MP,
                                              MP, MP, 32, 0.1767766953f, nullptr, 0);
  softmax_rows<<<dim3(KTOP, 8), 256, 0, stream>>>(Sb, MP, KTOP, (long long)MP * MP);
  // O_h = S_h V_h  (Kd padded to 512; V pad rows are zero)
  gemm_wmma<<<dim3(8, 8), 256, 0, stream>>>(Sb, MP, (long long)MP * MP, Vb, DD, 32,
                                            Ob, DD, 32, MP, 32, MP, 1.0f, nullptr, 0);
  gemm_wmma<<<dim3(64, 1), 256, 0, stream>>>(Ob, DD, 0, Wo, DD, 0, AttO, DD, 0,
                                             MP, DD, DD, 1.0f, nullptr, 0);
  ln_kernel<<<MP, 256, 0, stream>>>(AF, AttO, ln1g, ln1b, X1);

  // ---- FFN + residual LN -> final features (written straight to d_out) ----
  gemm_wmma<<<dim3(256, 1), 256, 0, stream>>>(X1, DD, 0, W1, 1024, 0, H1, 1024, 0,
                                              MP, 1024, DD, 1.0f, b1, 1);
  gemm_wmma<<<dim3(64, 1), 256, 0, stream>>>(H1, 1024, 0, W2, DD, 0, H2, DD, 0,
                                             MP, DD, 1024, 1.0f, b2, 0);
  ln_kernel<<<KTOP, 256, 0, stream>>>(X1, H2, ln2g, ln2b, out_final);

  // ---- scatter refined features, then the two fused conv heads ----
  scatter_kernel<<<KTOP, 128, 0, stream>>>(fmc, xs, ys, out_final);
  pack_w<<<288, 256, 0, stream>>>(whw1, WpkA);
  pack_w<<<288, 256, 0, stream>>>(orw1, WpkB);
  conv_fused<<<2048, 256, 0, stream>>>(fmc, WpkA, whb1, whw2, whb2, out_wh);
  conv_fused<<<2048, 256, 0, stream>>>(fmc, WpkB, orb1, orw2, orb2, out_ori);
}
